// AttentionHead_13202729468412
// MI455X (gfx1250) — matrix operand order repaired and checked by reading, not compiled, not run on hardware
//
#include <hip/hip_runtime.h>

#define DEVI __device__ __forceinline__

typedef __bf16 bf16;
typedef bf16  bf16x16 __attribute__((ext_vector_type(16)));
typedef float f32x8   __attribute__((ext_vector_type(8)));
typedef float f32x4   __attribute__((ext_vector_type(4)));
typedef unsigned int u32x4 __attribute__((ext_vector_type(4)));

static constexpr int B = 4, T = 4096, E = 1024, H = 64;
static constexpr float SCALE = 32.0f;  // sqrt(E) = sqrt(1024)

union BFrag { bf16x16 v; u32x4 q[2]; };

// A-frag style: two separate contiguous 8-element chunks
DEVI bf16x16 load_bf_frag2(const bf16* p0, const bf16* p1) {
  BFrag f;
  f.q[0] = *(const u32x4*)p0;
  f.q[1] = *(const u32x4*)p1;
  return f.v;
}
// B-frag style: two 8-element chunks at p and p + 16
DEVI bf16x16 load_bf_frag16(const bf16* p) {
  BFrag f;
  f.q[0] = *(const u32x4*)p;
  f.q[1] = *(const u32x4*)(p + 16);
  return f.v;
}

DEVI f32x8 wmma_bf16(bf16x16 a, bf16x16 b, f32x8 c) {
  return __builtin_amdgcn_wmma_f32_16x16x32_bf16(false, a, false, b, (short)0, c,
                                                 false, false);
}

// Row reductions across the 16 lanes of a half-wave (row of a C-frag)
DEVI float red_max16(float v) {
  v = fmaxf(v, __shfl_xor(v, 1));
  v = fmaxf(v, __shfl_xor(v, 2));
  v = fmaxf(v, __shfl_xor(v, 4));
  v = fmaxf(v, __shfl_xor(v, 8));
  return v;
}
DEVI float red_sum16(float v) {
  v += __shfl_xor(v, 1);
  v += __shfl_xor(v, 2);
  v += __shfl_xor(v, 4);
  v += __shfl_xor(v, 8);
  return v;
}

// ---------------------------------------------------------------------------
// Pass 0: one-time fp32 -> bf16 conversion of a weight matrix (H x E)
// ---------------------------------------------------------------------------
__global__ __launch_bounds__(256) void cvt_w_kernel(const float* __restrict__ w,
                                                    bf16* __restrict__ o) {
  const int i = (blockIdx.x * 256 + threadIdx.x) * 4;
  const f32x4 f = *(const f32x4*)(w + i);
#pragma unroll
  for (int t = 0; t < 4; ++t) o[i + t] = (bf16)f[t];
}

// ---------------------------------------------------------------------------
// Pass 1: q = x Wq^T, k = x Wk^T, v = x Wv^T  (x fp32, W pre-converted bf16)
// One block = one 16-row tile of T; the x tile is converted to bf16 once per
// block through a double-buffered LDS tile shared by all 4 waves. Wave w owns
// head-dims [16w, 16w+16). v is stored transposed: vT[b][h][t].
// ---------------------------------------------------------------------------
__global__ __launch_bounds__(128) void qkv_proj_kernel(
    const float* __restrict__ x, const bf16* __restrict__ wq,
    const bf16* __restrict__ wk, const bf16* __restrict__ wv,
    bf16* __restrict__ qo, bf16* __restrict__ ko, bf16* __restrict__ vto) {
  __shared__ __align__(16) bf16 sX[2][16 * 32];  // double-buffered x tile

  const int tid = threadIdx.x;
  const int lane = tid & 31;
  const int wave = tid >> 5;
  const int j = lane & 15;
  const int half = lane >> 4;
  const int tile = blockIdx.x;           // B*T/16 tiles
  const int b = tile >> 8;               // T/16 == 256
  const int t0 = (tile & 255) << 4;
  const int nb = wave << 4;              // head-dim tile base

  // staging role: thread -> (row, 4-col chunk) of the 16x32 x tile
  const int srow = tid >> 3;
  const int scol = (tid & 7) * 4;
  const float* xs = x + (size_t)(b * T + t0 + srow) * E + scol;

  const bf16* wqr = wq + (size_t)(nb + j) * E;
  const bf16* wkr = wk + (size_t)(nb + j) * E;
  const bf16* wvr = wv + (size_t)(nb + j) * E;

  f32x8 aq = {}, ak = {}, av = {};

  // prologue: stage K-chunk 0
  {
    const f32x4 f = *(const f32x4*)xs;
    bf16* d = &sX[0][srow * 32 + scol];
#pragma unroll
    for (int t = 0; t < 4; ++t) d[t] = (bf16)f[t];
  }
  __syncthreads();

  for (int it = 0; it < E / 32; ++it) {
    const int k0 = it * 32;
    if (it + 1 < E / 32) {  // stage next chunk into the other buffer
      __builtin_prefetch(xs + k0 + 64, 0, 1);
      const f32x4 f = *(const f32x4*)(xs + k0 + 32);
      bf16* d = &sX[(it + 1) & 1][srow * 32 + scol];
#pragma unroll
      for (int t = 0; t < 4; ++t) d[t] = (bf16)f[t];
    }
    const bf16* xb = sX[it & 1];
    const bf16x16 a =
        load_bf_frag2(xb + j * 32 + half * 8, xb + j * 32 + 16 + half * 8);
    const bf16x16 bq = load_bf_frag16(wqr + k0 + half * 8);
    const bf16x16 bk = load_bf_frag16(wkr + k0 + half * 8);
    const bf16x16 bv = load_bf_frag16(wvr + k0 + half * 8);
    aq = wmma_bf16(a, bq, aq);
    ak = wmma_bf16(a, bk, ak);
    av = wmma_bf16(a, bv, av);
    __syncthreads();
  }

#pragma unroll
  for (int r = 0; r < 8; ++r) {
    const int t = t0 + r + 8 * half;
    const size_t qidx = (size_t)(b * T + t) * H + nb + j;
    qo[qidx] = (bf16)aq[r];
    ko[qidx] = (bf16)ak[r];
    vto[(size_t)(b * H + nb + j) * T + t] = (bf16)av[r];
  }
}

// ---------------------------------------------------------------------------
// Pass 2: causal flash attention. One wave per 32-row query block (two 16-row
// sub-tiles), streaming 32-key tiles. Per key tile: 8 K/V B-frag loads feed
// 16 WMMAs (8 for S, 8 for P*V). Online softmax uses 16-lane shuffles; P is
// routed C-layout -> LDS -> A-layout (wave-local, LDS in-order).
// ---------------------------------------------------------------------------
__global__ __launch_bounds__(128) void flash_attn_kernel(
    const bf16* __restrict__ qw, const bf16* __restrict__ kw,
    const bf16* __restrict__ vtw, float* __restrict__ out) {
  __shared__ __align__(16) bf16 sP[4 * 16 * 32];  // per-wave 16x32 P tile

  const int lane = threadIdx.x & 31;
  const int wave = threadIdx.x >> 5;
  const int j = lane & 15;
  const int half = lane >> 4;
  const int qpair = blockIdx.x * 4 + wave;
  const int b = qpair >> 7;              // T/32 == 128 blocks per batch
  const int q0 = (qpair & 127) << 5;     // 32 query rows per wave

  const bf16* qb = qw  + (size_t)b * T * H;
  const bf16* kb = kw  + (size_t)b * T * H;
  const bf16* vb = vtw + (size_t)b * H * T;

  // Q A-frags: 2 query sub-tiles x 2 K-dim chunks
  bf16x16 qa[2][2];
#pragma unroll
  for (int qs = 0; qs < 2; ++qs) {
    const bf16* qrow = qb + (size_t)(q0 + qs * 16 + j) * H;
    qa[qs][0] = load_bf_frag2(qrow + half * 8,      qrow + 16 + half * 8);
    qa[qs][1] = load_bf_frag2(qrow + 32 + half * 8, qrow + 48 + half * 8);
  }

  float m[2][8], l[2][8];
  f32x8 o[2][4];
#pragma unroll
  for (int qs = 0; qs < 2; ++qs) {
#pragma unroll
    for (int r = 0; r < 8; ++r) { m[qs][r] = -1e30f; l[qs][r] = 0.0f; }
#pragma unroll
    for (int n = 0; n < 4; ++n) o[qs][n] = (f32x8){};
  }

  bf16* pl = sP + wave * (16 * 32);
  const int ntiles = (q0 + 63) >> 5;     // keys needed: [0, q0+32)

  for (int kt32 = 0; kt32 < ntiles; ++kt32) {
    const int kt = kt32 << 5;
    // K^T B-frags: key subtile {0,1} x K-dim chunk {0,1}
    const bf16* kr0 = kb + (size_t)(kt + j) * H;
    const bf16* kr1 = kb + (size_t)(kt + 16 + j) * H;
    const bf16x16 b00 = load_bf_frag16(kr0 + 0  + half * 8);
    const bf16x16 b01 = load_bf_frag16(kr0 + 32 + half * 8);
    const bf16x16 b10 = load_bf_frag16(kr1 + 0  + half * 8);
    const bf16x16 b11 = load_bf_frag16(kr1 + 32 + half * 8);

    // V B-frags (shared by both query sub-tiles)
    bf16x16 vf[4];
#pragma unroll
    for (int n = 0; n < 4; ++n)
      vf[n] = load_bf_frag16(vb + (size_t)(n * 16 + j) * T + kt + half * 8);

    // prefetch next key tile (K rows + V columns) into cache
    __builtin_prefetch(kr0 + 32 * H, 0, 1);
    __builtin_prefetch(kr1 + 32 * H, 0, 1);
    __builtin_prefetch(vb + (size_t)(2 * lane) * T + kt + 32, 0, 1);

    const int c0 = kt + j, c1 = kt + 16 + j;  // global key index per lane

#pragma unroll
    for (int qs = 0; qs < 2; ++qs) {
      const f32x8 z = {};
      f32x8 s0 = wmma_bf16(qa[qs][0], b00, z);
      s0 = wmma_bf16(qa[qs][1], b01, s0);
      f32x8 s1 = wmma_bf16(qa[qs][0], b10, z);
      s1 = wmma_bf16(qa[qs][1], b11, s1);

      float p0v[8], p1v[8], alpha[8];
#pragma unroll
      for (int r = 0; r < 8; ++r) {
        const int rowg = q0 + qs * 16 + r + 8 * half;  // global query row
        const float v0 = (c0 <= rowg) ? s0[r] * SCALE : -1e30f;
        const float v1 = (c1 <= rowg) ? s1[r] * SCALE : -1e30f;
        const float tmax = red_max16(fmaxf(v0, v1));
        const float mn = fmaxf(m[qs][r], tmax);
        alpha[r] = __expf(m[qs][r] - mn);
        m[qs][r] = mn;
        p0v[r] = __expf(v0 - mn);
        p1v[r] = __expf(v1 - mn);
        const float ts = red_sum16(p0v[r] + p1v[r]);
        l[qs][r] = l[qs][r] * alpha[r] + ts;
      }
#pragma unroll
      for (int n = 0; n < 4; ++n)
#pragma unroll
        for (int r = 0; r < 8; ++r) o[qs][n][r] *= alpha[r];

      // C-layout -> LDS (row-major 16x32). Wave-local: LDS ops from one wave
      // complete in order, so no barrier is needed around the round trip.
#pragma unroll
      for (int r = 0; r < 8; ++r) {
        pl[(r + 8 * half) * 32 + j]      = (bf16)p0v[r];
        pl[(r + 8 * half) * 32 + 16 + j] = (bf16)p1v[r];
      }
      const bf16x16 pa =
          load_bf_frag2(pl + j * 32 + half * 8, pl + j * 32 + 16 + half * 8);

#pragma unroll
      for (int n = 0; n < 4; ++n) o[qs][n] = wmma_bf16(pa, vf[n], o[qs][n]);
    }
  }

  float* ob = out + (size_t)b * T * H;
#pragma unroll
  for (int qs = 0; qs < 2; ++qs)
#pragma unroll
    for (int r = 0; r < 8; ++r) {
      const float inv = 1.0f / l[qs][r];
#pragma unroll
      for (int n = 0; n < 4; ++n)
        ob[(size_t)(q0 + qs * 16 + r + 8 * half) * H + n * 16 + j] =
            o[qs][n][r] * inv;
    }
}

extern "C" void kernel_launch(void* const* d_in, const int* in_sizes, int n_in,
                              void* d_out, int out_size, void* d_ws, size_t ws_size,
                              hipStream_t stream) {
  const float* x  = (const float*)d_in[0];
  const float* Wq = (const float*)d_in[1];
  const float* Wk = (const float*)d_in[2];
  const float* Wv = (const float*)d_in[3];
  float* out = (float*)d_out;

  const size_t BTH = (size_t)B * T * H;   // 1,048,576
  const size_t WHE = (size_t)H * E;       // 65,536
  bf16* qw  = (bf16*)d_ws;
  bf16* kw  = qw + BTH;
  bf16* vtw = kw + BTH;                   // transposed [B,H,T]
  bf16* wqb = vtw + BTH;
  bf16* wkb = wqb + WHE;
  bf16* wvb = wkb + WHE;

  cvt_w_kernel<<<WHE / 1024, 256, 0, stream>>>(Wq, wqb);
  cvt_w_kernel<<<WHE / 1024, 256, 0, stream>>>(Wk, wkb);
  cvt_w_kernel<<<WHE / 1024, 256, 0, stream>>>(Wv, wvb);
  qkv_proj_kernel<<<B * T / 16, 128, 0, stream>>>(x, wqb, wkb, wvb, qw, kw, vtw);
  flash_attn_kernel<<<B * T / 32 / 4, 128, 0, stream>>>(qw, kw, vtw, out);
}
